// RelGCN_41772851920989
// MI455X (gfx1250) — matrix operand order repaired and checked by reading, not compiled, not run on hardware
//
#include <hip/hip_runtime.h>
#include <math.h>

// ---------------------------------------------------------------------------
// RelGCN on MI455X (gfx1250, wave32).
// Heavy op: m = sum_r adj[r] @ H[r]  -> GEMM M=8192,N=64,K=24576 per layer.
// adj = 805MB f32/layer streamed from HBM (23.3 TB/s -> ~69us floor for 2 layers).
// AI = 32 F/B -> need ~745 TF/s effective; use v_wmma_f32_16x16x32_bf16 with a
// Dekker hi/lo bf16 split (3 WMMA terms) for ~fp32 accuracy at bf16-WMMA speed.
// adj converted f32->bf16(hi,lo) in registers: HBM traffic stays at f32 minimum.
// ---------------------------------------------------------------------------

typedef __attribute__((ext_vector_type(16))) __bf16 v16bf;
typedef __attribute__((ext_vector_type(8)))  float  v8f;
typedef __attribute__((ext_vector_type(4)))  float  v4f;

#define NN      8192
#define RR      3
#define DF      64
#define KSPL    2                 // K-split per relation
#define NCHUNK  (RR * KSPL)       // 6 partial-sum chunks
#define KCHUNK  (NN / KSPL)       // 4096

// Load 16 consecutive-K f32 of one A row into the 16-bit WMMA A-fragment
// element order and split into bf16 hi + residual lo.
// This code's A order (contiguous k, B matches): lanes 0-15 hold K 0..15, lanes 16-31 hold
// K 16..31 of row M = lane&15. elem e -> VGPR e/2 (ascending K).
static __device__ __forceinline__ void load_split_a(const float* __restrict__ rowk,
                                                    int lane_hi, v16bf& hi, v16bf& lo) {
  const float* p = rowk + (lane_hi ? 16 : 0);
  float v[16];
  *(v4f*)(v + 0)  = *(const v4f*)(p + 0);   // K run: kb .. kb+15
  *(v4f*)(v + 4)  = *(const v4f*)(p + 4);
  *(v4f*)(v + 8)  = *(const v4f*)(p + 8);
  *(v4f*)(v + 12) = *(const v4f*)(p + 12);
#pragma unroll
  for (int i = 0; i < 16; ++i) {
    float f  = v[i];
    __bf16 h = (__bf16)f;
    hi[i] = h;
    lo[i] = (__bf16)(f - (float)h);
  }
}

static __device__ __forceinline__ v8f wmma3(v16bf ahi, v16bf alo, v16bf bhi, v16bf blo, v8f c) {
  c = __builtin_amdgcn_wmma_f32_16x16x32_bf16(false, ahi, false, bhi, (short)0, c, false, false);
  c = __builtin_amdgcn_wmma_f32_16x16x32_bf16(false, alo, false, bhi, (short)0, c, false, false);
  c = __builtin_amdgcn_wmma_f32_16x16x32_bf16(false, ahi, false, blo, (short)0, c, false, false);
  return c;
}

// ---- transpose W [r][k][h] -> [r][h][k] and split to bf16 hi/lo planes ----
__global__ __launch_bounds__(256) void relgcn_prep_w(
    const float* __restrict__ W1, const float* __restrict__ W2,
    __bf16* __restrict__ wt1h, __bf16* __restrict__ wt1l,
    __bf16* __restrict__ wt2h, __bf16* __restrict__ wt2l) {
  int i = blockIdx.x * 256 + threadIdx.x;            // 0 .. 2*3*64*64-1
  int which = i / (RR * DF * DF);
  int j     = i % (RR * DF * DF);
  int r  = j / (DF * DF);
  int k  = (j / DF) % DF;
  int h  = j % DF;
  float f = (which ? W2 : W1)[j];
  __bf16 hh = (__bf16)f;
  __bf16 ll = (__bf16)(f - (float)hh);
  int o = (r * DF + h) * DF + k;
  if (which) { wt2h[o] = hh; wt2l[o] = ll; } else { wt1h[o] = hh; wt1l[o] = ll; }
}

// ---- H[r] = act @ W[r] + b[r]; store transposed bf16 hi/lo [r][h][m] ----
__global__ __launch_bounds__(256) void relgcn_linear(
    const float* __restrict__ act, const __bf16* __restrict__ wth,
    const __bf16* __restrict__ wtl, const float* __restrict__ bias,
    __bf16* __restrict__ hth, __bf16* __restrict__ htl) {
  const int lane = threadIdx.x & 31;
  const int wid  = threadIdx.x >> 5;
  const int lo16 = lane & 15;
  const int hi16 = lane >> 4;

  const int g  = blockIdx.x * 8 + wid;   // 1536 waves total
  const int r  = g >> 9;                 // / 512 row-blocks
  const int m0 = (g & 511) * 16;

  const float* arow = act + (size_t)(m0 + lo16) * DF;

  v8f acc[4];
#pragma unroll
  for (int t = 0; t < 4; ++t) {
    float bv = bias[r * DF + t * 16 + lo16];  // col depends only on lane
#pragma unroll
    for (int v = 0; v < 8; ++v) acc[t][v] = bv;
  }

#pragma unroll
  for (int k0 = 0; k0 < DF; k0 += 32) {
    v16bf ahi, alo;
    load_split_a(arow + k0, hi16, ahi, alo);
    const int kb2 = k0 + hi16 * 16;   // B: lanes 0-15 K 0..15, lanes 16-31 K 16..31
#pragma unroll
    for (int t = 0; t < 4; ++t) {
      const int brow = (r * DF + t * 16 + lo16) * DF + kb2;
      v16bf bh = *(const v16bf*)(wth + brow);
      v16bf bl = *(const v16bf*)(wtl + brow);
      acc[t] = wmma3(ahi, alo, bh, bl, acc[t]);
    }
  }

  // C/D layout: VGPR v, lane L -> row m0 + v + 8*(L>=16), col t*16 + (L&15)
#pragma unroll
  for (int t = 0; t < 4; ++t) {
#pragma unroll
    for (int v = 0; v < 8; ++v) {
      int m = m0 + v + 8 * hi16;
      int h = t * 16 + lo16;
      float f  = acc[t][v];
      __bf16 hh = (__bf16)f;
      size_t o = (size_t)(r * DF + h) * NN + m;
      hth[o] = hh;
      htl[o] = (__bf16)(f - (float)hh);
    }
  }
}

// ---- partial[c] = adj[r, :, kchunk] @ H[r][kchunk, :]  (f32 partials) ----
__global__ __launch_bounds__(256) void relgcn_adj_gemm(
    const float* __restrict__ adj, const __bf16* __restrict__ hth,
    const __bf16* __restrict__ htl, float* __restrict__ part) {
  const int lane = threadIdx.x & 31;
  const int wid  = threadIdx.x >> 5;
  const int lo16 = lane & 15;
  const int hi16 = lane >> 4;

  const int chunk = blockIdx.y;
  const int r     = chunk / KSPL;
  const int kbase = (chunk % KSPL) * KCHUNK;
  const int n0    = blockIdx.x * 128 + wid * 16;

  const float*  arow   = adj + ((size_t)r * NN + (n0 + lo16)) * NN;
  const __bf16* bbaseh = hth + (size_t)(r * DF) * NN;
  const __bf16* bbasel = htl + (size_t)(r * DF) * NN;

  v8f acc[4] = {};

  const int kend = kbase + KCHUNK;
  for (int k0 = kbase; k0 < kend; k0 += 32) {
    if (k0 + 1024 < kend) __builtin_prefetch(arow + k0 + 1024, 0, 1);
    v16bf ahi, alo;
    load_split_a(arow + k0, hi16, ahi, alo);
    const int kb2 = k0 + hi16 * 16;
#pragma unroll
    for (int t = 0; t < 4; ++t) {
      const size_t brow = (size_t)(t * 16 + lo16) * NN + kb2;
      v16bf bh = *(const v16bf*)(bbaseh + brow);
      v16bf bl = *(const v16bf*)(bbasel + brow);
      acc[t] = wmma3(ahi, alo, bh, bl, acc[t]);
    }
  }

  float* pout = part + (size_t)chunk * NN * DF;
#pragma unroll
  for (int t = 0; t < 4; ++t) {
#pragma unroll
    for (int v = 0; v < 8; ++v) {
      int n = n0 + v + 8 * hi16;
      int h = t * 16 + lo16;
      pout[(size_t)n * DF + h] = acc[t][v];  // lanes 0-15 contiguous in h
    }
  }
}

// ---- deterministic reduction over K-chunks + tanh ----
__global__ __launch_bounds__(256) void relgcn_reduce_tanh(
    const float* __restrict__ part, float* __restrict__ out) {
  int i = blockIdx.x * 256 + threadIdx.x;  // 0 .. NN*DF-1
  float s = 0.f;
#pragma unroll
  for (int c = 0; c < NCHUNK; ++c) s += part[(size_t)c * NN * DF + i];
  out[i] = tanhf(s);
}

extern "C" void kernel_launch(void* const* d_in, const int* in_sizes, int n_in,
                              void* d_out, int out_size, void* d_ws, size_t ws_size,
                              hipStream_t stream) {
  const float* x   = (const float*)d_in[0];
  const float* adj = (const float*)d_in[1];
  const float* W1  = (const float*)d_in[2];
  const float* b1  = (const float*)d_in[3];
  const float* W2  = (const float*)d_in[4];
  const float* b2  = (const float*)d_in[5];
  float* out = (float*)d_out;

  // workspace carve-out (~21.1 MB total)
  char* ws = (char*)d_ws;
  size_t cur = 0;
  auto carve = [&](size_t bytes) { size_t o = cur; cur = (cur + bytes + 255) & ~(size_t)255; return o; };
  const size_t WT_B  = (size_t)RR * DF * DF * sizeof(__bf16);       // 24 KB
  const size_t HT_B  = (size_t)RR * DF * NN * sizeof(__bf16);       // 3 MB
  const size_t PT_B  = (size_t)NCHUNK * NN * DF * sizeof(float);    // 12.6 MB
  const size_t ACT_B = (size_t)NN * DF * sizeof(float);             // 2 MB

  __bf16* wt1h = (__bf16*)(ws + carve(WT_B));
  __bf16* wt1l = (__bf16*)(ws + carve(WT_B));
  __bf16* wt2h = (__bf16*)(ws + carve(WT_B));
  __bf16* wt2l = (__bf16*)(ws + carve(WT_B));
  __bf16* hth  = (__bf16*)(ws + carve(HT_B));
  __bf16* htl  = (__bf16*)(ws + carve(HT_B));
  float*  part = (float*)(ws + carve(PT_B));
  float*  act1 = (float*)(ws + carve(ACT_B));

  const dim3 blk(256);
  // weight transpose + bf16 hi/lo split (both layers)
  relgcn_prep_w<<<(2 * RR * DF * DF) / 256, blk, 0, stream>>>(W1, W2, wt1h, wt1l, wt2h, wt2l);

  // ---- layer 1 ----
  relgcn_linear<<<(RR * (NN / 16)) / 8, blk, 0, stream>>>(x, wt1h, wt1l, b1, hth, htl);
  relgcn_adj_gemm<<<dim3(NN / 128, NCHUNK), blk, 0, stream>>>(adj, hth, htl, part);
  relgcn_reduce_tanh<<<(NN * DF) / 256, blk, 0, stream>>>(part, act1);

  // ---- layer 2 ----
  relgcn_linear<<<(RR * (NN / 16)) / 8, blk, 0, stream>>>(act1, wt2h, wt2l, b2, hth, htl);
  relgcn_adj_gemm<<<dim3(NN / 128, NCHUNK), blk, 0, stream>>>(adj, hth, htl, part);
  relgcn_reduce_tanh<<<(NN * DF) / 256, blk, 0, stream>>>(part, out);
}
